// DensePoseV1ConvXGNSparseHead_80410377716045
// MI455X (gfx1250) — matrix operand order repaired and checked by reading, not compiled, not run on hardware
//
#include <hip/hip_runtime.h>

typedef __attribute__((ext_vector_type(16))) __bf16 v16bf;
typedef __attribute__((ext_vector_type(8)))  float  v8f;

#define N_PTS  131072
#define C0     256
#define HID    512
#define KT     9
#define LAYERS 8
#define NT     32          // 512 / 16 column tiles

__device__ __forceinline__ unsigned short f2bf(float f) {
    unsigned u = __float_as_uint(f);
    unsigned r = (u + 0x7FFFu + ((u >> 16) & 1u)) >> 16;   // RNE
    return (unsigned short)r;
}

union AFrag { v16bf v; uint4 u[2]; };

// ---------------- feature fp32 -> bf16 ----------------
__global__ void cvt_feat(const float* __restrict__ in, unsigned short* __restrict__ out, size_t n) {
    for (size_t i = (size_t)blockIdx.x * blockDim.x + threadIdx.x; i < n;
         i += (size_t)gridDim.x * blockDim.x)
        out[i] = f2bf(in[i]);
}

// ---------------- pack weights [nmat][K][Cin][HID] fp32 -> B-fragment bf16 -----
// Layout: block (mat,k,ct,nt) holds 32x16 bf16; lane L (0..31) stores 16 bf16
// contiguously: n = nt*16 + (L&15), c = ct*32 + (L>>4)*8 + (j&7) + (j>>3)*16  (j = 0..15,
// VGPR j/2, low/high half-dword) -- matches 16-bit B 32x16 WMMA layout.
__global__ void pack_weights(const float* __restrict__ w, unsigned short* __restrict__ out,
                             int Cin, int nmat) {
    int CT = Cin >> 5;
    size_t total = (size_t)nmat * KT * CT * NT * 32;
    for (size_t id = (size_t)blockIdx.x * blockDim.x + threadIdx.x; id < total;
         id += (size_t)gridDim.x * blockDim.x) {
        int L = (int)(id & 31);
        size_t q = id >> 5;
        int nt = (int)(q % NT); q /= NT;
        int ct = (int)(q % CT); q /= CT;
        int k  = (int)(q % KT); q /= KT;
        int mat = (int)q;
        int n  = nt * 16 + (L & 15);
        int cb = ct * 32 + (L >> 4) * 8;
        const float* src = w + ((size_t)mat * KT + k) * Cin * HID;
        unsigned short* dst = out + (((((size_t)mat * KT + k) * CT + ct) * NT + nt) * 32 + L) * 16;
#pragma unroll
        for (int j = 0; j < 16; ++j)
            dst[j] = f2bf(src[(size_t)(cb + (j & 7) + ((j >> 3) << 4)) * HID + n]);
    }
}

// ---------------- fused gather -> WMMA GEMM -> GroupNorm -> ReLU --------------
__global__ __launch_bounds__(256) void subm_layer(
    const unsigned short* __restrict__ actin,  // [N, Cin] bf16
    const int*            __restrict__ nbr,    // [N, 9]
    const unsigned short* __restrict__ wpack,  // packed bf16 B fragments
    const float*          __restrict__ gamma,  // [512]
    const float*          __restrict__ beta,   // [512]
    unsigned short*       __restrict__ actout, // [N,512] bf16 (if !writeF32)
    float*                __restrict__ foutp,  // [N,512] fp32 (if writeF32)
    int Cin, int writeF32)
{
    extern __shared__ unsigned short lds[];    // [32][Cin + 8] bf16, padded stride
    const int CS   = Cin + 8;
    const int CT   = Cin >> 5;
    const int tid  = threadIdx.x;
    const int wave = tid >> 5;
    const int lane = tid & 31;
    const int hi   = lane >> 4;
    const int ln   = lane & 15;
    const int base = blockIdx.x * 32;

    v8f acc[2][4];
    const v8f vzero = {0.f, 0.f, 0.f, 0.f, 0.f, 0.f, 0.f, 0.f};
#pragma unroll
    for (int mi = 0; mi < 2; ++mi)
#pragma unroll
        for (int ni = 0; ni < 4; ++ni) acc[mi][ni] = vzero;

    for (int k = 0; k < KT; ++k) {
        __syncthreads();                       // LDS reads of previous tap done
        // cooperative gather: 32 rows x Cin bf16, 64-byte tasks
        int ntask = 32 * CT;
        for (int t = tid; t < ntask; t += 256) {
            int r  = t / CT;
            int cs = t % CT;
            int idx = nbr[(size_t)(base + r) * KT + k];
            uint4* dst = (uint4*)(lds + (size_t)r * CS + cs * 32);
            if (idx < 0) {
                uint4 z = {0u, 0u, 0u, 0u};
                dst[0] = z; dst[1] = z; dst[2] = z; dst[3] = z;
            } else {
                const uint4* src = (const uint4*)(actin + (size_t)idx * Cin + cs * 32);
                dst[0] = src[0]; dst[1] = src[1]; dst[2] = src[2]; dst[3] = src[3];
            }
        }
        __syncthreads();

        for (int ct = 0; ct < CT; ++ct) {
            // A fragments (16x32 bf16): lane half 'hi' holds K in
            // {hi*8..hi*8+7, 16+hi*8..16+hi*8+7} for row M = ln (+16 for mi=1)
            AFrag a0, a1;
            int ca = ct * 32 + hi * 8;
            a0.u[0] = *(const uint4*)(lds + (size_t)ln        * CS + ca);
            a0.u[1] = *(const uint4*)(lds + (size_t)ln        * CS + ca + 16);
            a1.u[0] = *(const uint4*)(lds + (size_t)(16 + ln) * CS + ca);
            a1.u[1] = *(const uint4*)(lds + (size_t)(16 + ln) * CS + ca + 16);
#pragma unroll
            for (int ni = 0; ni < 4; ++ni) {
                int nt = wave * 4 + ni;
                const uint4* bp = (const uint4*)(wpack +
                    ((((size_t)k * CT + ct) * NT + nt) * 32 + lane) * 16);
                AFrag b;
                b.u[0] = bp[0]; b.u[1] = bp[1];
                acc[0][ni] = __builtin_amdgcn_wmma_f32_16x16x32_bf16(
                    false, a0.v, false, b.v, (short)0, acc[0][ni], false, false);
                acc[1][ni] = __builtin_amdgcn_wmma_f32_16x16x32_bf16(
                    false, a1.v, false, b.v, (short)0, acc[1][ni], false, false);
            }
        }
    }

    // ---- GroupNorm (16-ch groups == one 16x16 tile) + ReLU, in registers ----
#pragma unroll
    for (int mi = 0; mi < 2; ++mi) {
#pragma unroll
        for (int ni = 0; ni < 4; ++ni) {
            int col = wave * 64 + ni * 16 + ln;
            float g  = gamma[col];
            float bb = beta[col];
#pragma unroll
            for (int r = 0; r < 8; ++r) {
                float v = acc[mi][ni][r];
                float s = v, ss = v * v;
#pragma unroll
                for (int m = 1; m <= 8; m <<= 1) {   // stays within half-wave
                    s  += __shfl_xor(s,  m, 32);
                    ss += __shfl_xor(ss, m, 32);
                }
                float mean = s * 0.0625f;
                float var  = ss * 0.0625f - mean * mean;
                float y = (v - mean) * rsqrtf(var + 1e-5f) * g + bb;
                y = fmaxf(y, 0.0f);
                int row = base + mi * 16 + r + hi * 8;
                if (writeF32) foutp[(size_t)row * HID + col] = y;
                else          actout[(size_t)row * HID + col] = f2bf(y);
            }
        }
    }
}

extern "C" void kernel_launch(void* const* d_in, const int* in_sizes, int n_in,
                              void* d_out, int out_size, void* d_ws, size_t ws_size,
                              hipStream_t stream) {
    const float* feat  = (const float*)d_in[0];   // [N,256]
    const int*   nbr   = (const int*)  d_in[1];   // [N,9]
    const float* w0    = (const float*)d_in[2];   // [9,256,512]
    const float* wrest = (const float*)d_in[3];   // [7,9,512,512]
    const float* gamma = (const float*)d_in[4];   // [8,512]
    const float* beta  = (const float*)d_in[5];   // [8,512]
    float* out = (float*)d_out;                   // [N,512]

    unsigned short* ws16 = (unsigned short*)d_ws;
    const size_t wp0_sz     = (size_t)KT * (C0 / 32)  * NT * 32 * 16;  // bf16 elems
    const size_t wpr_stride = (size_t)KT * (HID / 32) * NT * 32 * 16;
    unsigned short* wp0  = ws16;
    unsigned short* wpr  = wp0 + wp0_sz;
    unsigned short* buf1 = wpr + 7 * wpr_stride;
    unsigned short* buf2 = buf1 + (size_t)N_PTS * HID;
    unsigned short* featb = buf2;   // [N,256] bf16, dead after layer0 -> alias buf2

    cvt_feat<<<1024, 256, 0, stream>>>(feat, featb, (size_t)N_PTS * C0);
    pack_weights<<<512,  256, 0, stream>>>(w0,    wp0, C0,  1);
    pack_weights<<<2048, 256, 0, stream>>>(wrest, wpr, HID, 7);

    const int blocks = N_PTS / 32;
    // layer 0: Cin = 256
    subm_layer<<<blocks, 256, 32 * (C0 + 8) * 2, stream>>>(
        featb, nbr, wp0, gamma, beta, buf1, nullptr, C0, 0);

    const unsigned short* cin = buf1;
    unsigned short*       cout = buf2;
    for (int l = 1; l < LAYERS; ++l) {
        int last = (l == LAYERS - 1);
        subm_layer<<<blocks, 256, 32 * (HID + 8) * 2, stream>>>(
            cin, nbr, wpr + (size_t)(l - 1) * wpr_stride,
            gamma + (size_t)l * HID, beta + (size_t)l * HID,
            last ? nullptr : cout, last ? out : nullptr, HID, last);
        const unsigned short* t = cin;
        cin = cout;
        cout = (unsigned short*)t;
    }
}
